// LongformerSelfAttention_67903432950239
// MI455X (gfx1250) — matrix operand order repaired and checked by reading, not compiled, not run on hardware
//
#include <hip/hip_runtime.h>
#include <hip/hip_bf16.h>
#include <stdint.h>
#include <stddef.h>

// ---------------------------------------------------------------------------
// Longformer sliding-window self-attention for MI455X (gfx1250, wave32, WMMA)
// B=2 S=4096 E=768 H=12 HD=64 W=256  band = 2W+1 = 513
// ---------------------------------------------------------------------------

typedef __bf16 bf16_t;
typedef bf16_t v16bf __attribute__((ext_vector_type(16)));
typedef bf16_t v8bf  __attribute__((ext_vector_type(8)));
typedef float  v8f   __attribute__((ext_vector_type(8)));

#define S_LEN   4096
#define EMB     768
#define NHEAD   12
#define HDIM    64
#define WWIN    256
#define BATCH   2
#define BAND    513
#define SUBQ    64            // queries handled per attention block
#define KSLAB   576           // SUBQ + 2*WWIN keys staged per block

// LDS row strides padded +8 bf16 (16B) so fragment row-walks are bank-conflict
// free: dword stride 36 (and 292 == 36 mod 64); 16 lanes x stride-36 covers 16
// distinct bank groups of the 64 LDS banks.
#define QK_STR  72            // qs/ks row stride (elements)
#define SC_STR  584           // sc/pf/vt row stride (elements)

// ---------------- gfx1250-specific helpers ----------------

__device__ __forceinline__ uint32_t lds_off_of(const void* p) {
  return (uint32_t)(uintptr_t)p;   // LDS generic addresses carry the DS offset in low 32 bits
}

// Async DMA: global -> LDS, 16 bytes per lane, tracked by ASYNCcnt.
__device__ __forceinline__ void async_copy_b128(void* ldsDst, const void* gSrc) {
  asm volatile("global_load_async_to_lds_b128 %0, %1, off"
               :: "v"(lds_off_of(ldsDst)), "v"((uint64_t)(uintptr_t)gSrc)
               : "memory");
}

__device__ __forceinline__ void wait_async0() {
  asm volatile("s_wait_asynccnt 0" ::: "memory");
}

__device__ __forceinline__ v8f wmma_bf16(v16bf a, v16bf b, v8f c) {
  // D = A(16x32 bf16) * B(32x16 bf16) + C(16x16 f32)
  return __builtin_amdgcn_wmma_f32_16x16x32_bf16(
      /*neg_a=*/false, a, /*neg_b=*/false, b,
      /*c_mod=*/(short)0, c, /*reuse_a=*/false, /*reuse_b=*/false);
}

// A-fragment (16x32 bf16) from row-major storage, rowStride in elements.
// lane<16 : M = rbase+lane,    elems 0..7 = K+0..7,  8..15 = K+16..23
// lane>=16: M = rbase+lane-16, elems 0..7 = K+8..15, 8..15 = K+24..31
__device__ __forceinline__ v16bf load_a_frag(const bf16_t* src, int rowStride,
                                             int rbase, int k0, int lane) {
  int r  = rbase + (lane & 15);
  int kk = k0 + ((lane >> 4) & 1) * 8;
  const bf16_t* p = src + (size_t)r * rowStride + kk;
  v8bf lo = *(const v8bf*)p;
  v8bf hi = *(const v8bf*)(p + 16);
  return __builtin_shufflevector(lo, hi, 0,1,2,3,4,5,6,7,8,9,10,11,12,13,14,15);
}

// B-fragment (32x16 bf16), B[k][n] stored n-major / K-contiguous:
// element B[k0+e][nbase+n] lives at src[(nbase+n)*rowStride + k0 + e]
// lane<16 : N = nbase+lane,    elems 0..7 = K+0..7,  8..15 = K+16..23
// lane>=16: N = nbase+lane-16, elems 0..7 = K+8..15, 8..15 = K+24..31
__device__ __forceinline__ v16bf load_b_frag(const bf16_t* src, int rowStride,
                                             int nbase, int k0, int lane) {
  int n  = nbase + (lane & 15);
  int kk = k0 + ((lane >> 4) & 1) * 8;
  const bf16_t* p = src + (size_t)n * rowStride + kk;
  v8bf lo = *(const v8bf*)p;
  v8bf hi = *(const v8bf*)(p + 16);
  return __builtin_shufflevector(lo, hi, 0,1,2,3,4,5,6,7,8,9,10,11,12,13,14,15);
}

// ---------------- conversion kernels ----------------

__global__ void cvt_x(const float* __restrict__ x, bf16_t* __restrict__ xb, int n) {
  int i = (blockIdx.x * blockDim.x + threadIdx.x) * 4;
  if (i + 3 < n) {
    float4 f = *(const float4*)(x + i);
    xb[i+0] = (bf16_t)f.x; xb[i+1] = (bf16_t)f.y;
    xb[i+2] = (bf16_t)f.z; xb[i+3] = (bf16_t)f.w;
  }
}

// Wt[n][k] = W{n/768}[k][n%768]   (n in [0,2304): q|k|v concatenated)
__global__ void cvt_w(const float* __restrict__ Wq, const float* __restrict__ Wk,
                      const float* __restrict__ Wv, bf16_t* __restrict__ Wt) {
  int i = blockIdx.x * blockDim.x + threadIdx.x;
  if (i >= 3 * EMB * EMB) return;
  int n = i / EMB, k = i - (i / EMB) * EMB;
  int proj = n / EMB;
  int e = n - proj * EMB;
  const float* src = (proj == 0) ? Wq : (proj == 1 ? Wk : Wv);
  Wt[(size_t)n * EMB + k] = (bf16_t)src[(size_t)k * EMB + e];
}

// ---------------- fused QKV projection GEMM (bf16 WMMA) ----------------
// C[8192 x 2304] = X[8192 x 768] * W[768 x 2304]; epilogue: +bias, ReLU,
// 1/sqrt(HD) for Q, scatter to per-head [B*H][S][64] bf16 buffers.

__global__ __launch_bounds__(256) void qkv_gemm(
    const bf16_t* __restrict__ Xb, const bf16_t* __restrict__ Wt,
    const float* __restrict__ bq, const float* __restrict__ bk,
    const float* __restrict__ bv,
    bf16_t* __restrict__ Qb, bf16_t* __restrict__ Kb, bf16_t* __restrict__ Vb) {
  const int tid  = threadIdx.x;
  const int lane = tid & 31;
  const int wave = tid >> 5;
  const int bm = blockIdx.x % 128;          // 64-row block tiles over M=8192
  const int bn = blockIdx.x / 128;          // 128-col block tiles over N=2304
  const int mBase = bm * 64 + (wave & 3) * 16;
  const int nBase = bn * 128 + (wave >> 2) * 64;

  v8f acc[4] = {{}, {}, {}, {}};
  for (int k0 = 0; k0 < EMB; k0 += 32) {
    __builtin_prefetch(Xb + (size_t)(mBase + (lane & 15)) * EMB + k0 + 64, 0, 1);
    v16bf a = load_a_frag(Xb, EMB, mBase, k0, lane);
#pragma unroll
    for (int t = 0; t < 4; ++t) {
      v16bf b = load_b_frag(Wt, EMB, nBase + t * 16, k0, lane);
      acc[t] = wmma_bf16(a, b, acc[t]);
    }
  }

  const int mRow0 = mBase + ((lane >> 4) & 1) * 8;
#pragma unroll
  for (int t = 0; t < 4; ++t) {
    // A 16-wide tile never straddles a projection (768) or head (64) boundary,
    // so decompose the wave-uniform tile base, not the per-lane column.
    const int nTile = nBase + t * 16;
    const int proj  = nTile / EMB;
    const int e0    = nTile - proj * EMB;
    const int h     = e0 >> 6;
    const int d     = (e0 & 63) + (lane & 15);
    const float* bias  = (proj == 0) ? bq : (proj == 1 ? bk : bv);
    bf16_t*      dstBf = (proj == 0) ? Qb : (proj == 1 ? Kb : Vb);
    const float bval = bias[e0 + (lane & 15)];
    const float scl  = (proj == 0) ? 0.125f : 1.0f;   // 1/sqrt(64)
#pragma unroll
    for (int r = 0; r < 8; ++r) {
      int m = mRow0 + r;
      int bi = m >> 12, s = m & (S_LEN - 1);
      float v = acc[t][r] + bval;
      v = v > 0.f ? v : 0.f;
      v *= scl;
      dstBf[(((size_t)bi * NHEAD + h) * S_LEN + s) * HDIM + d] = (bf16_t)v;
    }
  }
}

// ---------------- banded attention (per head, chunk, 64-query sub-chunk) ----
// LDS map (dynamic, 316416 B):
//   [0,      9216)  qs  [64][72]  bf16   (phase 1-2; reused as pf later)
//   [9216,  92160)  ks  [576][72] bf16   (phase 1-2; reused as pf later)
//   [0,     74752)  pf  [64][584] bf16   (overlay of qs+ks)
//   [92160,241664)  sc  [64][584] f32
//   [241664,316416) vt  [64][584] bf16   (v transposed: vt[d][y])

__global__ __launch_bounds__(256) void band_attn(
    const bf16_t* __restrict__ Qb, const bf16_t* __restrict__ Kb,
    const bf16_t* __restrict__ Vb, const int* __restrict__ am,
    float* __restrict__ attnOut, float* __restrict__ probsOut) {
  extern __shared__ char smem[];
  bf16_t* qs = (bf16_t*)smem;
  bf16_t* ks = (bf16_t*)(smem + 9216);
  bf16_t* pf = (bf16_t*)smem;
  float*  sc = (float*)(smem + 92160);
  bf16_t* vt = (bf16_t*)(smem + 241664);
  __shared__ float redMax[256];
  __shared__ float redSum[256];

  const int tid = threadIdx.x, lane = tid & 31, wave = tid >> 5;
  const int blk = blockIdx.x;
  const int sub = blk & 3;
  const int c   = (blk >> 2) & 15;
  const int bh  = blk >> 6;                 // 0..23
  const int b   = bh / NHEAD, h = bh - b * NHEAD;
  const int qStart = c * WWIN + sub * SUBQ;
  const int slab0  = qStart - WWIN;         // absolute key index of ks row 0
  const bf16_t* Qbh = Qb + (size_t)bh * S_LEN * HDIM;
  const bf16_t* Kbh = Kb + (size_t)bh * S_LEN * HDIM;
  const bf16_t* Vbh = Vb + (size_t)bh * S_LEN * HDIM;

  // -- Phase 1: stage q/k via async DMA, v transposed via regs --------------
  for (int i = tid; i < SUBQ * 8; i += 256) {             // q: 64 rows x 8 chunks
    int row = i >> 3, part = i & 7;
    async_copy_b128(qs + row * QK_STR + part * 8,
                    Qbh + (size_t)(qStart + row) * HDIM + part * 8);
  }
  for (int i = tid; i < KSLAB * 8; i += 256) {            // k: 576 rows x 8 chunks
    int row = i >> 3, part = i & 7;
    int ka = slab0 + row;
    if (ka >= 0 && ka < S_LEN)
      async_copy_b128(ks + row * QK_STR + part * 8,
                      Kbh + (size_t)ka * HDIM + part * 8);
    else
      *(v8bf*)(ks + row * QK_STR + part * 8) = (v8bf){};
  }
  for (int i = tid; i < KSLAB * 8; i += 256) {            // v -> vt[d][y]
    int row = i >> 3, part = i & 7;
    int ka = slab0 + row;
    if (ka >= 0 && ka < S_LEN) {
      v8bf g = *(const v8bf*)(Vbh + (size_t)ka * HDIM + part * 8);
#pragma unroll
      for (int e = 0; e < 8; ++e) vt[(part * 8 + e) * SC_STR + row] = g[e];
    } else {
#pragma unroll
      for (int e = 0; e < 8; ++e) vt[(part * 8 + e) * SC_STR + row] = (bf16_t)0.0f;
    }
  }
  wait_async0();
  __syncthreads();

  // -- Phase 2: scores = q (64xHD) . k^T (HDx576) via WMMA ------------------
  {
    const int mTile = wave >> 1, half = wave & 1;
    v16bf aFrag0 = load_a_frag(qs, QK_STR, mTile * 16, 0, lane);
    v16bf aFrag1 = load_a_frag(qs, QK_STR, mTile * 16, 32, lane);
    for (int nt = 0; nt < 18; ++nt) {
      int n0 = (half * 18 + nt) * 16;
      v8f acc = {};
      acc = wmma_bf16(aFrag0, load_b_frag(ks, QK_STR, n0, 0, lane), acc);
      acc = wmma_bf16(aFrag1, load_b_frag(ks, QK_STR, n0, 32, lane), acc);
      int n  = n0 + (lane & 15);
      int mr = mTile * 16 + ((lane >> 4) & 1) * 8;
#pragma unroll
      for (int r = 0; r < 8; ++r) sc[(mr + r) * SC_STR + n] = acc[r];
    }
  }
  __syncthreads();

  // -- Phase 3: zero pf, band row-max ---------------------------------------
  {
    v8bf z = {};
    for (int i = tid; i < SUBQ * SC_STR / 8; i += 256) *((v8bf*)pf + i) = z;
  }
  const int r = tid >> 2, q4 = tid & 3;
  const int sAbs = qStart + r;
  float lmax = -3.0e38f;
  for (int j = q4; j < BAND; j += 4) {
    int ka = sAbs - WWIN + j;
    float v;
    if (ka >= 0 && ka < S_LEN)
      v = sc[r * SC_STR + r + j] + (am[b * S_LEN + ka] != 0 ? -10000.0f : 0.0f);
    else
      v = -1.0e9f;
    lmax = fmaxf(lmax, v);
  }
  redMax[tid] = lmax;
  __syncthreads();
  const float mx = fmaxf(fmaxf(redMax[r * 4 + 0], redMax[r * 4 + 1]),
                         fmaxf(redMax[r * 4 + 2], redMax[r * 4 + 3]));

  // -- Phase 4: exp, fill pf band (bf16, unnormalized), partial sums --------
  float lsum = 0.f;
  for (int j = q4; j < BAND; j += 4) {
    int ka = sAbs - WWIN + j;
    float v;
    if (ka >= 0 && ka < S_LEN)
      v = sc[r * SC_STR + r + j] + (am[b * S_LEN + ka] != 0 ? -10000.0f : 0.0f);
    else
      v = -1.0e9f;
    float e = __expf(v - mx);
    lsum += e;
    pf[r * SC_STR + r + j] = (bf16_t)e;
  }
  redSum[tid] = lsum;
  __syncthreads();
  const float rowSumMine =
      redSum[r * 4 + 0] + redSum[r * 4 + 1] + redSum[r * 4 + 2] + redSum[r * 4 + 3];
  const float invMine = 1.0f / rowSumMine;

  // -- Phase 5: attn = pf (64x576) . v (576xHD) via WMMA, normalized --------
  {
    const int mTile = wave >> 1, half = wave & 1;
    v8f acc2[2] = {{}, {}};
    for (int k0 = 0; k0 < KSLAB; k0 += 32) {
      v16bf a = load_a_frag(pf, SC_STR, mTile * 16, k0, lane);
#pragma unroll
      for (int t = 0; t < 2; ++t) {
        v16bf bb = load_b_frag(vt, SC_STR, (half * 2 + t) * 16, k0, lane);
        acc2[t] = wmma_bf16(a, bb, acc2[t]);
      }
    }
    const int mr0 = mTile * 16 + ((lane >> 4) & 1) * 8;
#pragma unroll
    for (int t = 0; t < 2; ++t) {
      int d = (half * 2 + t) * 16 + (lane & 15);
      int e = h * HDIM + d;
#pragma unroll
      for (int rr = 0; rr < 8; ++rr) {
        int row = mr0 + rr;
        float rs = redSum[row * 4 + 0] + redSum[row * 4 + 1] +
                   redSum[row * 4 + 2] + redSum[row * 4 + 3];
        float val = acc2[t][rr] / rs;
        int s = qStart + row;
        attnOut[((size_t)b * S_LEN + s) * EMB + e] = val;
      }
    }
  }

  // -- Phase 6: normalized fp32 probs straight to d_out ---------------------
  for (int j = q4; j < BAND; j += 4) {
    int ka = sAbs - WWIN + j;
    float v;
    if (ka >= 0 && ka < S_LEN)
      v = sc[r * SC_STR + r + j] + (am[b * S_LEN + ka] != 0 ? -10000.0f : 0.0f);
    else
      v = -1.0e9f;
    float p = __expf(v - mx) * invMine;
    probsOut[(((size_t)b * S_LEN + sAbs) * NHEAD + h) * BAND + j] = p;
  }
}

// ---------------- host launcher ----------------

extern "C" void kernel_launch(void* const* d_in, const int* in_sizes, int n_in,
                              void* d_out, int out_size, void* d_ws, size_t ws_size,
                              hipStream_t stream) {
  const float* X   = (const float*)d_in[0];
  const int*   am  = (const int*)d_in[1];
  const float* Wq  = (const float*)d_in[2];
  const float* bq  = (const float*)d_in[3];
  const float* Wk  = (const float*)d_in[4];
  const float* bk  = (const float*)d_in[5];
  const float* Wv  = (const float*)d_in[6];
  const float* bvp = (const float*)d_in[7];

  char* ws = (char*)d_ws;
  bf16_t* Xb = (bf16_t*)(ws + 0);              // 8192*768*2      = 12,582,912
  bf16_t* Wt = (bf16_t*)(ws + 12582912);       // 2304*768*2      =  3,538,944
  bf16_t* Qb = (bf16_t*)(ws + 16121856);       // 24*4096*64*2    = 12,582,912
  bf16_t* Kb = (bf16_t*)(ws + 28704768);
  bf16_t* Vb = (bf16_t*)(ws + 41287680);       // ends at 53,870,592 bytes

  float* attnOut  = (float*)d_out;                          // B*S*E
  float* probsOut = attnOut + (size_t)BATCH * S_LEN * EMB;  // B*S*H*513

  cvt_x<<<(BATCH * S_LEN * EMB) / (256 * 4), 256, 0, stream>>>(
      X, Xb, BATCH * S_LEN * EMB);
  cvt_w<<<(3 * EMB * EMB + 255) / 256, 256, 0, stream>>>(Wq, Wk, Wv, Wt);
  qkv_gemm<<<128 * 18, 256, 0, stream>>>(Xb, Wt, bq, bk, bvp, Qb, Kb, Vb);
  band_attn<<<BATCH * NHEAD * 16 * 4, 256, 316416, stream>>>(
      Qb, Kb, Vb, am, attnOut, probsOut);
}
